// DalleSelfAttention_73375221285582
// MI455X (gfx1250) — matrix operand order repaired and checked by reading, not compiled, not run on hardware
//
#include <hip/hip_runtime.h>
#include <cstdint>

#define H_DIM    2048
#define NHEAD    16
#define HEAD_DIM 128
#define BATCH    2
#define SEQ      2048
#define QSCALE   0.088388347648318447f   /* 1/sqrt(128) */

typedef __attribute__((ext_vector_type(16))) _Float16 v16h;
typedef __attribute__((ext_vector_type(8)))  _Float16 v8h;
typedef __attribute__((ext_vector_type(8)))  float    v8f;

#define WMMA(a, b, c) __builtin_amdgcn_wmma_f32_16x16x32_f16( \
    false, (a), false, (b), (short)0, (c), false, false)

// ---------------- fragment loaders (overloaded on element type) ----------------
// A (16x32 f16): lane l<16 = row M=l, elems 0..15 = K k0..k0+15 (hi lanes +16).
__device__ __forceinline__ v16h load_a(const float* __restrict__ p0, int ld, int lane) {
    const int l = lane & 15, hi = lane >> 4;
    const float* p = p0 + (size_t)l * ld + hi * 16;
    float4 f0 = *(const float4*)(p);
    float4 f1 = *(const float4*)(p + 4);
    float4 f2 = *(const float4*)(p + 8);
    float4 f3 = *(const float4*)(p + 12);
    v16h v;
    v[0]=(_Float16)f0.x; v[1]=(_Float16)f0.y; v[2]=(_Float16)f0.z; v[3]=(_Float16)f0.w;
    v[4]=(_Float16)f1.x; v[5]=(_Float16)f1.y; v[6]=(_Float16)f1.z; v[7]=(_Float16)f1.w;
    v[8]=(_Float16)f2.x; v[9]=(_Float16)f2.y; v[10]=(_Float16)f2.z; v[11]=(_Float16)f2.w;
    v[12]=(_Float16)f3.x; v[13]=(_Float16)f3.y; v[14]=(_Float16)f3.z; v[15]=(_Float16)f3.w;
    return v;
}
__device__ __forceinline__ v16h load_a(const _Float16* __restrict__ p0, int ld, int lane) {
    const int l = lane & 15, hi = lane >> 4;
    const _Float16* p = p0 + (size_t)l * ld + hi * 16;
    v8h a = *(const v8h*)(p);
    v8h b = *(const v8h*)(p + 8);
    v16h v;
#pragma unroll
    for (int i = 0; i < 8; ++i) { v[i] = a[i]; v[8 + i] = b[i]; }
    return v;
}
// B (32x16 f16): lane l<16 = col N=l, elems 0..15 = K k0..k0+15 (hi lanes +16).
__device__ __forceinline__ v16h load_b(const float* __restrict__ p0, int ld, int lane) {
    const int l = lane & 15, hi = lane >> 4;
    const float* p = p0 + (size_t)l * ld + hi * 16;
    float4 f0 = *(const float4*)(p);
    float4 f1 = *(const float4*)(p + 4);
    float4 f2 = *(const float4*)(p + 8);
    float4 f3 = *(const float4*)(p + 12);
    v16h v;
    v[0]=(_Float16)f0.x; v[1]=(_Float16)f0.y; v[2]=(_Float16)f0.z; v[3]=(_Float16)f0.w;
    v[4]=(_Float16)f1.x; v[5]=(_Float16)f1.y; v[6]=(_Float16)f1.z; v[7]=(_Float16)f1.w;
    v[8]=(_Float16)f2.x; v[9]=(_Float16)f2.y; v[10]=(_Float16)f2.z; v[11]=(_Float16)f2.w;
    v[12]=(_Float16)f3.x; v[13]=(_Float16)f3.y; v[14]=(_Float16)f3.z; v[15]=(_Float16)f3.w;
    return v;
}
__device__ __forceinline__ v16h load_b(const _Float16* __restrict__ p0, int ld, int lane) {
    const int l = lane & 15, hi = lane >> 4;
    const _Float16* p = p0 + (size_t)l * ld + hi * 16;
    v8h a = *(const v8h*)(p);
    v8h b = *(const v8h*)(p + 8);
    v16h v;
#pragma unroll
    for (int i = 0; i < 8; ++i) { v[i] = a[i]; v[8 + i] = b[i]; }
    return v;
}
// per-lane row prefetch (-> global_prefetch_b8)
template <typename T>
__device__ __forceinline__ void pf_row(const T* p, int ld, int lane) {
    __builtin_prefetch(p + (size_t)(lane & 15) * ld, 0, 3);
}
// async global -> LDS copy, 16B per lane (GLOBAL_LOAD_ASYNC_TO_LDS_B128, ASYNCcnt)
__device__ __forceinline__ void async_b128(uint32_t lds_off, const void* gaddr) {
    asm volatile("global_load_async_to_lds_b128 %0, %1, off"
                 :: "v"(lds_off), "v"(gaddr) : "memory");
}

// ---------------- Kernel 0: bulk f32 -> f16 conversion ----------------
__global__ __launch_bounds__(256) void cvt_f32_f16(
    const float* __restrict__ in, _Float16* __restrict__ out, int n) {
    const int i = (blockIdx.x * 256 + threadIdx.x) * 8;
    if (i >= n) return;
    float4 a = *(const float4*)(in + i);
    float4 b = *(const float4*)(in + i + 4);
    v8h v;
    v[0]=(_Float16)a.x; v[1]=(_Float16)a.y; v[2]=(_Float16)a.z; v[3]=(_Float16)a.w;
    v[4]=(_Float16)b.x; v[5]=(_Float16)b.y; v[6]=(_Float16)b.z; v[7]=(_Float16)b.w;
    *(v8h*)(out + i) = v;
}

// ---------------- Kernel 1: QKV projection ----------------
template <typename TA, typename TB>
__global__ __launch_bounds__(128) void qkv_gemm(
    const TA* __restrict__ hid, const TB* __restrict__ w,
    const float* __restrict__ bqkv,
    _Float16* __restrict__ qb, _Float16* __restrict__ kb, _Float16* __restrict__ vtb) {
    const int lane = threadIdx.x & 31;
    const int wave = threadIdx.x >> 5;
    const int lid = lane & 15;
    const int hi8 = (lane >> 4) << 3;
    const int mbase = blockIdx.x * 128 + (wave >> 1) * 64;
    const int nbase = blockIdx.y * 128 + (wave & 1) * 64;

    const v8f zero = {0.f, 0.f, 0.f, 0.f, 0.f, 0.f, 0.f, 0.f};
    v8f acc[4][4];
#pragma unroll
    for (int i = 0; i < 4; ++i)
#pragma unroll
        for (int j = 0; j < 4; ++j) acc[i][j] = zero;

#pragma unroll 1
    for (int k0 = 0; k0 < H_DIM; k0 += 32) {
        v16h a[4], bb[4];
#pragma unroll
        for (int i = 0; i < 4; ++i)
            a[i] = load_a(hid + (size_t)(mbase + 16 * i) * H_DIM + k0, H_DIM, lane);
#pragma unroll
        for (int j = 0; j < 4; ++j)
            bb[j] = load_b(w + (size_t)(nbase + 16 * j) * H_DIM + k0, H_DIM, lane);
        if (k0 + 96 < H_DIM) {
#pragma unroll
            for (int i = 0; i < 4; ++i)
                pf_row(hid + (size_t)(mbase + 16 * i) * H_DIM + k0 + 64, H_DIM, lane);
#pragma unroll
            for (int j = 0; j < 4; ++j)
                pf_row(w + (size_t)(nbase + 16 * j) * H_DIM + k0 + 64, H_DIM, lane);
        }
#pragma unroll
        for (int i = 0; i < 4; ++i)
#pragma unroll
            for (int j = 0; j < 4; ++j) acc[i][j] = WMMA(a[i], bb[j], acc[i][j]);
    }

#pragma unroll
    for (int i = 0; i < 4; ++i) {
#pragma unroll
        for (int j = 0; j < 4; ++j) {
            const int n0 = nbase + 16 * j;
            const int ncol = n0 + lid;
            const float bias = bqkv[ncol];
            const int seg = n0 / H_DIM;            // 0=Q 1=K 2=V (uniform per tile)
            const int nf = ncol - seg * H_DIM;
            const int head = nf / HEAD_DIM;
            const int d = nf % HEAD_DIM;
#pragma unroll
            for (int v = 0; v < 8; ++v) {
                const int mrow = mbase + 16 * i + v + hi8;
                const int bidx = mrow / SEQ;
                const int stok = mrow - bidx * SEQ;
                const float val = acc[i][j][v] + bias;
                if (seg == 0)
                    qb[((size_t)(bidx * NHEAD + head) * SEQ + stok) * HEAD_DIM + d] =
                        (_Float16)(val * QSCALE);
                else if (seg == 1)
                    kb[((size_t)(bidx * NHEAD + head) * SEQ + stok) * HEAD_DIM + d] =
                        (_Float16)val;
                else
                    vtb[((size_t)(bidx * NHEAD + head) * HEAD_DIM + d) * SEQ + stok] =
                        (_Float16)val;
            }
        }
    }
}

// ---------------- Kernel 2: causal flash attention (transposed, LDS-staged) ----
// 4 waves share one (b, head, 64-query block). Per 32-key block, the workgroup
// async-DMAs K (32x128, 8KB) and V^T (128x32, 8KB) to LDS once (double-buffered),
// then each wave runs S^T = K·Q^T and ctx^T += V^T·P^T from ds_load fragments.
#define KV_BUF_BYTES 16384u   /* 8KB K + 8KB V per buffer */

__global__ __launch_bounds__(128) void attn_kernel(
    const _Float16* __restrict__ qb, const _Float16* __restrict__ kb,
    const _Float16* __restrict__ vtb, _Float16* __restrict__ ctx) {
    extern __shared__ __align__(16) char smem[];
    const int tid = threadIdx.x;
    const int lane = tid & 31;
    const int wave = tid >> 5;
    const int lid = lane & 15;
    const int hi = lane >> 4;
    const int hi8 = hi << 3;

    int blk = blockIdx.x;
    const int QB = SEQ / 64;
    const int qblk = blk % QB; blk /= QB;
    const int head = blk % NHEAD;
    const int b = blk / NHEAD;
    const int qbase = qblk * 64 + wave * 16;

    const _Float16* Qh = qb + (size_t)(b * NHEAD + head) * SEQ * HEAD_DIM;
    const _Float16* Kh = kb + (size_t)(b * NHEAD + head) * SEQ * HEAD_DIM;
    const _Float16* Vh = vtb + (size_t)(b * NHEAD + head) * HEAD_DIM * SEQ;

    v16h qf[4];
#pragma unroll
    for (int t = 0; t < 4; ++t)
        qf[t] = load_b(Qh + (size_t)qbase * HEAD_DIM + 32 * t, HEAD_DIM, lane);

    const v8f zero = {0.f, 0.f, 0.f, 0.f, 0.f, 0.f, 0.f, 0.f};
    v8f o[8];
#pragma unroll
    for (int t = 0; t < 8; ++t) o[t] = zero;
    float m = -1e30f, l = 0.f;
    const int q = qbase + lid;
    // all 4 waves loop to the block-max trip count (wave 3); compute is guarded.
    const int nkb = (qblk * 64 + 48 + 16 + 31) >> 5;

    // stage one 32-key K/V block into LDS buffer `bufOff` (8 async b128 per thread)
    auto stage = [&](int kbase, uint32_t bufOff) {
        const char* gK = (const char*)(Kh + (size_t)kbase * HEAD_DIM);
#pragma unroll
        for (int r = 0; r < 4; ++r)
            async_b128(bufOff + tid * 16 + r * 2048, gK + tid * 16 + r * 2048);
        const int d0 = tid >> 2;
        const int sg = (tid & 3) * 16;
#pragma unroll
        for (int r = 0; r < 4; ++r) {
            const int d = d0 + r * 32;
            async_b128(bufOff + 8192 + d * 64 + sg,
                       (const char*)(Vh + (size_t)d * SEQ + kbase) + sg);
        }
    };

    stage(0, 0u);
#pragma unroll 1
    for (int kbI = 0; kbI < nkb; ++kbI) {
        const int kbase = kbI * 32;
        const uint32_t buf = (kbI & 1) ? KV_BUF_BYTES : 0u;
        if (kbI + 1 < nkb) {
            stage(kbase + 32, (kbI & 1) ? 0u : KV_BUF_BYTES);
            asm volatile("s_wait_asynccnt 0x8" ::: "memory");
        } else {
            asm volatile("s_wait_asynccnt 0x0" ::: "memory");
        }
        __syncthreads();

        if (kbase <= qbase + 15) {      // wave-uniform causal guard
            const _Float16* Kt = (const _Float16*)(smem + buf);          // 32 x 128
            const _Float16* Vt = (const _Float16*)(smem + buf + 8192);   // 128 x 32
            v8f s0 = zero, s1 = zero;
#pragma unroll
            for (int t = 0; t < 4; ++t) {
                v16h ka = load_a(Kt + 32 * t, HEAD_DIM, lane);
                s0 = WMMA(ka, qf[t], s0);
            }
#pragma unroll
            for (int t = 0; t < 4; ++t) {
                v16h ka = load_a(Kt + 16 * HEAD_DIM + 32 * t, HEAD_DIM, lane);
                s1 = WMMA(ka, qf[t], s1);
            }
            // causal mask + online softmax (per-lane = per-query column)
            float tm = -1e30f;
#pragma unroll
            for (int j = 0; j < 8; ++j) {
                const int k0i = kbase + hi8 + j;
                if (k0i > q) s0[j] = -1e30f;
                if (k0i + 16 > q) s1[j] = -1e30f;
                tm = fmaxf(tm, fmaxf(s0[j], s1[j]));
            }
            tm = fmaxf(tm, __shfl_xor(tm, 16, 32));
            const float mnew = fmaxf(m, tm);
            const float sc = __expf(m - mnew);
            float rs = 0.f;
#pragma unroll
            for (int j = 0; j < 8; ++j) {
                s0[j] = __expf(s0[j] - mnew);
                s1[j] = __expf(s1[j] - mnew);
                rs += s0[j] + s1[j];
            }
            rs += __shfl_xor(rs, 16, 32);
            l = l * sc + rs;
            m = mnew;
#pragma unroll
            for (int t = 0; t < 8; ++t)
#pragma unroll
                for (int j = 0; j < 8; ++j) o[t][j] *= sc;
            // P^T as B-fragment via one lane-16 exchange
            v16h pb;
#pragma unroll
            for (int j = 0; j < 8; ++j) {
                const float x0 = __shfl_xor(s0[j], 16, 32);
                const float x1 = __shfl_xor(s1[j], 16, 32);
                pb[j]     = (_Float16)(hi ? x1 : s0[j]);
                pb[8 + j] = (_Float16)(hi ? s1[j] : x0);
            }
#pragma unroll
            for (int t = 0; t < 8; ++t) {
                v16h va = load_a(Vt + (size_t)(16 * t) * 32, 32, lane);
                o[t] = WMMA(va, pb, o[t]);
            }
        }
        __syncthreads();   // everyone done reading `buf` before it is re-staged
    }

    const float linv = 1.f / l;
    const size_t rowoff = ((size_t)b * SEQ + q) * H_DIM + (size_t)head * HEAD_DIM;
#pragma unroll
    for (int t = 0; t < 8; ++t)
#pragma unroll
        for (int j = 0; j < 8; ++j)
            ctx[rowoff + 16 * t + j + hi8] = (_Float16)(o[t][j] * linv);
}

// ---------------- Kernel 3: output projection ----------------
template <typename TB>
__global__ __launch_bounds__(128) void out_gemm(
    const _Float16* __restrict__ ctx, const TB* __restrict__ w,
    const float* __restrict__ bout, float* __restrict__ out) {
    const int lane = threadIdx.x & 31;
    const int wave = threadIdx.x >> 5;
    const int lid = lane & 15;
    const int hi8 = (lane >> 4) << 3;
    const int mbase = blockIdx.x * 128 + (wave >> 1) * 64;
    const int nbase = blockIdx.y * 128 + (wave & 1) * 64;

    const v8f zero = {0.f, 0.f, 0.f, 0.f, 0.f, 0.f, 0.f, 0.f};
    v8f acc[4][4];
#pragma unroll
    for (int i = 0; i < 4; ++i)
#pragma unroll
        for (int j = 0; j < 4; ++j) acc[i][j] = zero;

#pragma unroll 1
    for (int k0 = 0; k0 < H_DIM; k0 += 32) {
        v16h a[4], bb[4];
#pragma unroll
        for (int i = 0; i < 4; ++i)
            a[i] = load_a(ctx + (size_t)(mbase + 16 * i) * H_DIM + k0, H_DIM, lane);
#pragma unroll
        for (int j = 0; j < 4; ++j)
            bb[j] = load_b(w + (size_t)(nbase + 16 * j) * H_DIM + k0, H_DIM, lane);
        if (k0 + 96 < H_DIM) {
#pragma unroll
            for (int i = 0; i < 4; ++i)
                pf_row(ctx + (size_t)(mbase + 16 * i) * H_DIM + k0 + 64, H_DIM, lane);
#pragma unroll
            for (int j = 0; j < 4; ++j)
                pf_row(w + (size_t)(nbase + 16 * j) * H_DIM + k0 + 64, H_DIM, lane);
        }
#pragma unroll
        for (int i = 0; i < 4; ++i)
#pragma unroll
            for (int j = 0; j < 4; ++j) acc[i][j] = WMMA(a[i], bb[j], acc[i][j]);
    }

#pragma unroll
    for (int i = 0; i < 4; ++i)
#pragma unroll
        for (int j = 0; j < 4; ++j) {
            const int ncol = nbase + 16 * j + lid;
            const float bias = bout[ncol];
#pragma unroll
            for (int v = 0; v < 8; ++v) {
                const int mrow = mbase + 16 * i + v + hi8;
                out[(size_t)mrow * H_DIM + ncol] = acc[i][j][v] + bias;
            }
        }
}

extern "C" void kernel_launch(void* const* d_in, const int* in_sizes, int n_in,
                              void* d_out, int out_size, void* d_ws, size_t ws_size,
                              hipStream_t stream) {
    const float* hid  = (const float*)d_in[0];
    // d_in[1] = ltor_mask: causality handled analytically, unused
    const float* wqkv = (const float*)d_in[2];
    const float* bqkv = (const float*)d_in[3];
    const float* wout = (const float*)d_in[4];
    const float* bout = (const float*)d_in[5];
    float* out = (float*)d_out;

    const size_t PER    = (size_t)BATCH * NHEAD * SEQ * HEAD_DIM; // 8,388,608
    const size_t HID_E  = (size_t)BATCH * SEQ * H_DIM;            // 8,388,608
    const size_t WQKV_E = (size_t)3 * H_DIM * H_DIM;              // 12,582,912
    const size_t WOUT_E = (size_t)H_DIM * H_DIM;                  //  4,194,304

    _Float16* qb   = (_Float16*)d_ws;
    _Float16* kb   = qb + PER;
    _Float16* vtb  = kb + PER;
    _Float16* ctx  = vtb + PER;
    _Float16* hid6 = ctx + PER;
    _Float16* wq6  = hid6 + HID_E;
    _Float16* wo6  = wq6 + WQKV_E;

    const size_t needed = (4 * PER + HID_E + WQKV_E + WOUT_E) * sizeof(_Float16);
    const unsigned attnGrid = BATCH * NHEAD * (SEQ / 64);
    const unsigned attnLds = 2 * KV_BUF_BYTES;   // 32 KB double buffer

    if (ws_size >= needed) {
        // one-time f32->f16 conversion, then pure-f16 GEMM inner loops
        cvt_f32_f16<<<dim3((unsigned)(HID_E / 2048)), 256, 0, stream>>>(hid, hid6, (int)HID_E);
        cvt_f32_f16<<<dim3((unsigned)(WQKV_E / 2048)), 256, 0, stream>>>(wqkv, wq6, (int)WQKV_E);
        cvt_f32_f16<<<dim3((unsigned)(WOUT_E / 2048)), 256, 0, stream>>>(wout, wo6, (int)WOUT_E);
        qkv_gemm<_Float16, _Float16><<<dim3(32, 48), 128, 0, stream>>>(
            hid6, wq6, bqkv, qb, kb, vtb);
        attn_kernel<<<dim3(attnGrid), 128, attnLds, stream>>>(qb, kb, vtb, ctx);
        out_gemm<_Float16><<<dim3(32, 16), 128, 0, stream>>>(ctx, wo6, bout, out);
    } else {
        // fallback: convert in the fragment loaders
        qkv_gemm<float, float><<<dim3(32, 48), 128, 0, stream>>>(
            hid, wqkv, bqkv, qb, kb, vtb);
        attn_kernel<<<dim3(attnGrid), 128, attnLds, stream>>>(qb, kb, vtb, ctx);
        out_gemm<float><<<dim3(32, 16), 128, 0, stream>>>(ctx, wout, bout, out);
    }
}
